// BilstmModel_28767690949359
// MI455X (gfx1250) — matrix operand order repaired and checked by reading, not compiled, not run on hardware
//
#include <hip/hip_runtime.h>

typedef unsigned short ushort_t;
typedef __attribute__((ext_vector_type(16))) __bf16 bf16x16;
typedef __attribute__((ext_vector_type(8)))  float  f32x8;
typedef __attribute__((ext_vector_type(4)))  unsigned u32x4;
typedef __attribute__((ext_vector_type(8)))  int      i32x8;
typedef __attribute__((ext_vector_type(4)))  int      i32x4;

// ---------- constants ----------
#define BB   64      // batch
#define SS   512     // seq len
#define DD   300     // input dim
#define HH   200     // hidden
#define NG   800     // 4*H
#define KWP  320     // D padded to mult of 32
#define KUP  224     // H padded to mult of 32
#define SA_STRIDE 328  // LDS A stride (elements, mult of 8)

// workspace byte offsets (256-aligned)
#define WT_OFF   0ull                       // bf16 [2][800][320]  = 1,024,000 B
#define UT_OFF   1024000ull                 // bf16 [2][800][224]  =   716,800 B
#define MASK_OFF 1740800ull                 // i32  [64*512]       =   131,072 B
#define XZ_OFF   1871872ull                 // f32  [2][512][64][800] = 209,715,200 B
#define HS_OFF   211587072ull               // f32  [2][512][64][200] =  52,428,800 B

// ---------- helpers ----------
__device__ __forceinline__ unsigned f2bf_bits(float f) {
    unsigned u = __builtin_bit_cast(unsigned, f);
    return (u + 0x7FFFu + ((u >> 16) & 1u)) >> 16;
}
__device__ __forceinline__ ushort_t f2bf_u(float f) { return (ushort_t)f2bf_bits(f); }
__device__ __forceinline__ unsigned pack2bf(float lo, float hi) {
    return f2bf_bits(lo) | (f2bf_bits(hi) << 16);
}

struct Bytes32 { uint4 lo, hi; };

__device__ __forceinline__ bf16x16 make_frag(uint4 lo, uint4 hi) {
    Bytes32 b; b.lo = lo; b.hi = hi;
    return __builtin_bit_cast(bf16x16, b);
}

// A-fragment (16x32 bf16, M x K): lane<16 -> K kb..kb+15 (kb=k0),
// lane>=16 -> kb=k0+16. Two 16B chunks at [row][kb] and [row][kb+8].
__device__ __forceinline__ bf16x16 load_fragA(const ushort_t* base, int stride, int row, int kb) {
    const uint4* p0 = (const uint4*)(base + row * stride + kb);
    const uint4* p1 = (const uint4*)(base + row * stride + kb + 8);
    return make_frag(*p0, *p1);
}

// B-fragment (32x16 bf16, K x N), B stored transposed [n][k]:
// lane n=lane&15, kb = k0 + (lane>=16 ? 16 : 0): 16 contiguous bf16 = 32B.
__device__ __forceinline__ bf16x16 load_fragB16(const ushort_t* p) {
    const uint4* q = (const uint4*)p;
    return make_frag(q[0], q[1]);
}

// ---------- Tensor Data Mover: 1-D tile load, global -> LDS ----------
// Descriptor per CDNA5 ISA:
//   g0: [1:0]=count=1, [63:32]=lds_addr, [120:64]=global_addr, [127:126]=type=2
//   g1: [17:16]=data_size, [20]=pad_enable, [24:22]=pad_interval, [31:25]=pad_amount,
//       [79:48]=tensor_dim0, [111:80]=tensor_dim1(=1), [127:112]=tile_dim0,
//       [207:160]=tensor_dim0_stride
__device__ __forceinline__ void tdm_load_1d(unsigned lds_off, const void* gsrc,
                                            unsigned nelem, unsigned dsz_code,
                                            unsigned pad_flags) {
    unsigned long long ga = (unsigned long long)(size_t)gsrc;
    u32x4 g0;
    g0[0] = 1u;                                   // count=1, user mode
    g0[1] = lds_off;                              // LDS byte address
    g0[2] = (unsigned)ga;                         // global_addr[31:0]
    g0[3] = (unsigned)(ga >> 32) | 0x80000000u;   // global_addr[56:32] | type=2
    i32x8 g1;
    g1[0] = (int)((dsz_code << 16) | pad_flags);  // data_size + pad controls
    g1[1] = (int)((nelem & 0xFFFFu) << 16);       // tensor_dim0[15:0] at bits 63:48
    g1[2] = (int)((nelem >> 16) | (1u << 16));    // tensor_dim0[31:16] | tensor_dim1=1
    g1[3] = (int)((nelem & 0xFFFFu) << 16);       // tile_dim0 at bits 127:112
    g1[4] = 0;                                    // tile_dim1=0, tile_dim2=0
    g1[5] = (int)nelem;                           // tensor_dim0_stride[31:0]
    g1[6] = 0;
    g1[7] = 0;
    i32x4 z4; z4[0] = 0; z4[1] = 0; z4[2] = 0; z4[3] = 0;
#if defined(__clang_major__) && (__clang_major__ >= 23)
    i32x8 z8;
    z8[0] = 0; z8[1] = 0; z8[2] = 0; z8[3] = 0;
    z8[4] = 0; z8[5] = 0; z8[6] = 0; z8[7] = 0;
    __builtin_amdgcn_tensor_load_to_lds(g0, g1, z4, z4, z8, 0);
#else
    __builtin_amdgcn_tensor_load_to_lds(g0, g1, z4, z4, 0);
#endif
}

// grouped (TDM-padded) LDS layout for the GEMM B tile:
// 64 bf16 elements (32 DWORDs = 128B) per group + 16B pad -> 144B group stride.
// row n = 5 groups (320 elems); element (n,k) -> byte (n*5 + k/64)*144 + (k%64)*2
__device__ __forceinline__ const ushort_t* sb_addr(const ushort_t* sB, int nn, int kf) {
    return (const ushort_t*)((const char*)sB + (nn * 5 + (kf >> 6)) * 144 + ((kf & 63) << 1));
}

// ---------- weight convert: W [300][800] -> Wt bf16 [dir][800][320] (transposed, K-padded) ----------
__global__ void kConvW(const float* __restrict__ Wf, const float* __restrict__ Wb,
                       ushort_t* __restrict__ Wt) {
    int idx = blockIdx.x * 256 + threadIdx.x;
    if (idx >= 2 * NG * KWP) return;
    int dir = idx / (NG * KWP);
    int rem = idx - dir * NG * KWP;
    int n = rem / KWP, k = rem - n * KWP;
    const float* W = dir ? Wb : Wf;
    float v = (k < DD) ? W[k * NG + n] : 0.0f;
    Wt[idx] = f2bf_u(v);
}

// ---------- U [200][800] -> Ut bf16 [dir][800][224] ----------
__global__ void kConvU(const float* __restrict__ Uf, const float* __restrict__ Ub,
                       ushort_t* __restrict__ Ut) {
    int idx = blockIdx.x * 256 + threadIdx.x;
    if (idx >= 2 * NG * KUP) return;
    int dir = idx / (NG * KUP);
    int rem = idx - dir * NG * KUP;
    int n = rem / KUP, k = rem - n * KUP;
    const float* U = dir ? Ub : Uf;
    float v = (k < HH) ? U[k * NG + n] : 0.0f;
    Ut[idx] = f2bf_u(v);
}

// ---------- mask[b*S+s] = any(x[b,s,:] != 0) ; one wave per row ----------
__global__ void kMask(const float* __restrict__ x, int* __restrict__ mask) {
    int row  = blockIdx.x * 8 + (threadIdx.x >> 5);
    int lane = threadIdx.x & 31;
    const float* xr = x + (size_t)row * DD;
    int any = 0;
    for (int d = lane; d < DD; d += 32) any |= (xr[d] != 0.0f);
    unsigned long long bal = __ballot(any);
    if (lane == 0) mask[row] = (bal != 0ull) ? 1 : 0;
}

// ---------- input GEMM: xz[dir][s][b][n] = x[b,s,:]@W[dir] + bias[dir] ----------
// grid (512, 5, 2), block 256 (8 waves: 4 M-waves x 2 N-waves), tile 64(M=batch) x 160(N)
// B tile is DMA'd by the Tensor Data Mover (with LDS padding) while all threads
// convert the A tile fp32->bf16 with vectorized b128 loads/stores.
__global__ void kInputGemm(const float* __restrict__ x, const ushort_t* __restrict__ Wt,
                           const float* __restrict__ bf, const float* __restrict__ bb,
                           float* __restrict__ xz) {
    extern __shared__ char smem_raw[];
    ushort_t* sA = (ushort_t*)smem_raw;                          // 64 x 328        = 41,984 B
    ushort_t* sB = (ushort_t*)(smem_raw + 64 * SA_STRIDE * 2);   // 800 groups x144 = 115,200 B

    const int s    = blockIdx.x;
    const int nblk = blockIdx.y * 160;
    const int dir  = blockIdx.z;
    const int tid  = threadIdx.x;

    // async: TDM copies 160x320 bf16 of Wt into LDS with 16B pad per 128B
    if (tid < 32) {
        const ushort_t* src = Wt + ((size_t)dir * NG + nblk) * KWP;
        tdm_load_1d((unsigned)(size_t)(void*)sB, src, 160 * KWP, 1u /*2B*/,
                    (1u << 20) | (4u << 22) | (3u << 25));  // pad_en, every 32 DW, +4 DW
    }

    // A: rows b=0..63 of x at timestep s, fp32 -> bf16, 8-element vector chunks.
    // 41 chunks per row: j<37 full, j==37 tail (296..299 + zeros), j>37 zero pad.
    for (int c = tid; c < 64 * 41; c += 256) {
        int m = c / 41, j = c - m * 41;
        int d0 = j * 8;
        uint4 v;
        if (j < 37) {
            const float* xr = x + ((size_t)m * SS + s) * DD + d0;
            __builtin_prefetch(xr + 64, 0, 1);
            float4 a = *(const float4*)(xr);
            float4 b = *(const float4*)(xr + 4);
            v.x = pack2bf(a.x, a.y); v.y = pack2bf(a.z, a.w);
            v.z = pack2bf(b.x, b.y); v.w = pack2bf(b.z, b.w);
        } else if (j == 37) {
            const float* xr = x + ((size_t)m * SS + s) * DD + 296;
            float4 a = *(const float4*)(xr);
            v.x = pack2bf(a.x, a.y); v.y = pack2bf(a.z, a.w);
            v.z = 0u; v.w = 0u;
        } else {
            v.x = v.y = v.z = v.w = 0u;
        }
        *(uint4*)(sA + m * SA_STRIDE + d0) = v;
    }
    if (tid < 32) __builtin_amdgcn_s_wait_tensorcnt(0);
    __syncthreads();

    const int wave = tid >> 5, lane = tid & 31;
    const int wm = wave >> 1, wn = wave & 1;
    const int m0   = wm * 16;
    const int nrel = wn * 80;
    const int l15  = lane & 15, hi = lane >> 4;

    f32x8 acc[5];
    #pragma unroll
    for (int t = 0; t < 5; t++)
        #pragma unroll
        for (int r = 0; r < 8; r++) acc[t][r] = 0.0f;

    #pragma unroll
    for (int k0 = 0; k0 < KWP; k0 += 32) {
        bf16x16 a = load_fragA(sA, SA_STRIDE, m0 + l15, k0 + hi * 16);
        const int kf = k0 + hi * 16;
        #pragma unroll
        for (int t = 0; t < 5; t++) {
            bf16x16 b = load_fragB16(sb_addr(sB, nrel + t * 16 + l15, kf));
            acc[t] = __builtin_amdgcn_wmma_f32_16x16x32_bf16(false, a, false, b,
                                                             (short)0, acc[t], false, false);
        }
    }

    const float* bias = dir ? bb : bf;
    #pragma unroll
    for (int t = 0; t < 5; t++) {
        int n = nblk + nrel + t * 16 + l15;
        float bv = bias[n];
        #pragma unroll
        for (int r = 0; r < 8; r++) {
            int b = m0 + r + 8 * hi;
            xz[(((size_t)dir * SS + s) * BB + b) * NG + n] = acc[t][r] + bv;
        }
    }
}

// ---------- persistent recurrent kernel ----------
// grid (4 batch-tiles, 2 dirs), block 320 (10 waves, each owns 80 z-columns).
// Per-step xz tile (16x800 f32, contiguous) is double-buffered in LDS via TDM,
// overlapped one step ahead; completion via TENSORcnt (in-order).
__global__ void kRecurrent(const float* __restrict__ xz, const ushort_t* __restrict__ Ut,
                           const int* __restrict__ mask, float* __restrict__ hseq) {
    extern __shared__ char smem_raw[];
    ushort_t* hbf  = (ushort_t*)smem_raw;                             // 16 x 224 bf16
    float*    zbuf = (float*)(smem_raw + 16 * KUP * 2);               // 16 x 800 f32
    float*    xzb0 = (float*)(smem_raw + 16 * KUP * 2 + 16 * NG * 4); // 16 x 800 f32
    float*    xzb1 = (float*)(smem_raw + 16 * KUP * 2 + 32 * NG * 4); // 16 x 800 f32

    const int bt  = blockIdx.x;
    const int dir = blockIdx.y;
    const int b0  = bt * 16;
    const int tid = threadIdx.x;
    const int wave = tid >> 5, lane = tid & 31, l15 = lane & 15, hi = lane >> 4;
    const int n0 = wave * 80;

    for (int i = tid; i < 16 * KUP; i += 320) hbf[i] = 0;
    float hstate[16], cstate[16];
    #pragma unroll
    for (int m = 0; m < 16; m++) { hstate[m] = 0.0f; cstate[m] = 0.0f; }

    // prologue: DMA step-0 tile into buffer 0
    if (tid < 32) {
        int s0 = dir ? (SS - 1) : 0;
        tdm_load_1d((unsigned)(size_t)(void*)xzb0,
                    xz + (((size_t)dir * SS + s0) * BB + b0) * NG,
                    16 * NG, 2u /*4B*/, 0u);
    }
    __syncthreads();

    const ushort_t* Ud = Ut + (size_t)dir * NG * KUP;

    for (int step = 0; step < SS; ++step) {
        const int s = dir ? (SS - 1 - step) : step;
        float* cur = (step & 1) ? xzb1 : xzb0;
        float* nxt = (step & 1) ? xzb0 : xzb1;

        if (tid < 32) {
            if (step + 1 < SS) {
                int sn = dir ? (SS - 2 - step) : (step + 1);
                tdm_load_1d((unsigned)(size_t)(void*)nxt,
                            xz + (((size_t)dir * SS + sn) * BB + b0) * NG,
                            16 * NG, 2u, 0u);
                __builtin_amdgcn_s_wait_tensorcnt(1);  // current-step tile done
            } else {
                __builtin_amdgcn_s_wait_tensorcnt(0);
            }
        }
        __syncthreads();

        // init accumulators from LDS-staged xz tile
        f32x8 acc[5];
        #pragma unroll
        for (int t = 0; t < 5; t++) {
            int n = n0 + t * 16 + l15;
            #pragma unroll
            for (int r = 0; r < 8; r++)
                acc[t][r] = cur[(r + 8 * hi) * NG + n];
        }

        // z += h @ U   (h: 16x224 bf16 in LDS, U: streamed from L2, transposed [n][k])
        #pragma unroll
        for (int k0 = 0; k0 < KUP; k0 += 32) {
            bf16x16 a = load_fragA(hbf, KUP, l15, k0 + hi * 16);
            #pragma unroll
            for (int t = 0; t < 5; t++) {
                const ushort_t* bp = Ud + (size_t)(n0 + t * 16 + l15) * KUP + k0 + hi * 16;
                bf16x16 bfrag = load_fragB16(bp);
                acc[t] = __builtin_amdgcn_wmma_f32_16x16x32_bf16(false, a, false, bfrag,
                                                                 (short)0, acc[t], false, false);
            }
        }

        // exchange z through LDS
        #pragma unroll
        for (int t = 0; t < 5; t++) {
            int n = n0 + t * 16 + l15;
            #pragma unroll
            for (int r = 0; r < 8; r++)
                zbuf[(r + 8 * hi) * NG + n] = acc[t][r];
        }
        __syncthreads();

        // gate nonlinearities + state update; threads 0..199 each own one h column
        if (tid < HH) {
            const int hh = tid;
            #pragma unroll
            for (int m = 0; m < 16; m++) {
                float zi = zbuf[m * NG + hh];
                float zf = zbuf[m * NG + HH + hh];
                float zg = zbuf[m * NG + 2 * HH + hh];
                float zo = zbuf[m * NG + 3 * HH + hh];
                float ig = 1.0f / (1.0f + __expf(-zi));
                float fg = 1.0f / (1.0f + __expf(-zf));
                float eg = __expf(2.0f * zg);
                float gg = (eg - 1.0f) / (eg + 1.0f);
                float og = 1.0f / (1.0f + __expf(-zo));
                float cn = fg * cstate[m] + ig * gg;
                float e2 = __expf(2.0f * cn);
                float hn = og * ((e2 - 1.0f) / (e2 + 1.0f));
                int mk = mask[(b0 + m) * SS + s];
                if (!mk) { cn = cstate[m]; hn = hstate[m]; }
                cstate[m] = cn; hstate[m] = hn;
                hbf[m * KUP + hh] = f2bf_u(hn);
                hseq[(((size_t)dir * SS + s) * BB + (b0 + m)) * HH + hh] = hn;
            }
        }
        __syncthreads();
    }
}

// ---------- attention pooling: one block per batch row ----------
__global__ void kAtt(const float* __restrict__ hseq, const float* __restrict__ watt,
                     float* __restrict__ out) {
    __shared__ float sc[SS];
    __shared__ float red[256];
    const int b = blockIdx.x, tid = threadIdx.x;
    const float* hf = hseq;
    const float* hb = hseq + (size_t)SS * BB * HH;

    for (int s = tid; s < SS; s += 256) {
        const float* pf = hf + ((size_t)s * BB + b) * HH;
        const float* pb = hb + ((size_t)s * BB + b) * HH;
        float acc = 0.0f;
        for (int k = 0; k < HH; k++) acc += watt[k] * pf[k];
        for (int k = 0; k < HH; k++) acc += watt[HH + k] * pb[k];
        sc[s] = acc;
    }
    __syncthreads();

    float mx = -3.4e38f;
    for (int s = tid; s < SS; s += 256) mx = fmaxf(mx, sc[s]);
    red[tid] = mx; __syncthreads();
    for (int off = 128; off > 0; off >>= 1) {
        if (tid < off) red[tid] = fmaxf(red[tid], red[tid + off]);
        __syncthreads();
    }
    mx = red[0]; __syncthreads();

    float lsum = 0.0f;
    for (int s = tid; s < SS; s += 256) { float e = __expf(sc[s] - mx); sc[s] = e; lsum += e; }
    red[tid] = lsum; __syncthreads();
    for (int off = 128; off > 0; off >>= 1) {
        if (tid < off) red[tid] += red[tid + off];
        __syncthreads();
    }
    float inv = 1.0f / red[0];
    for (int s = tid; s < SS; s += 256) sc[s] *= inv;
    __syncthreads();

    for (int k = tid; k < 2 * HH; k += 256) {
        const float* base = (k < HH) ? (hf + (size_t)b * HH + k)
                                     : (hb + (size_t)b * HH + (k - HH));
        float acc = 0.0f;
        for (int s = 0; s < SS; s++) acc += sc[s] * base[(size_t)s * BB * HH];
        out[b * 2 * HH + k] = acc;
    }
}

// ---------- launch ----------
extern "C" void kernel_launch(void* const* d_in, const int* in_sizes, int n_in,
                              void* d_out, int out_size, void* d_ws, size_t ws_size,
                              hipStream_t stream) {
    const float* x    = (const float*)d_in[0];
    const float* Wf   = (const float*)d_in[1];
    const float* Uf   = (const float*)d_in[2];
    const float* bf   = (const float*)d_in[3];
    const float* Wb   = (const float*)d_in[4];
    const float* Ub   = (const float*)d_in[5];
    const float* bb   = (const float*)d_in[6];
    const float* watt = (const float*)d_in[7];

    char* ws = (char*)d_ws;
    ushort_t* Wt   = (ushort_t*)(ws + WT_OFF);
    ushort_t* Ut   = (ushort_t*)(ws + UT_OFF);
    int*      mask = (int*)     (ws + MASK_OFF);
    float*    xz   = (float*)   (ws + XZ_OFF);
    float*    hseq = (float*)   (ws + HS_OFF);
    float*    out  = (float*)d_out;

    kConvW<<<(2 * NG * KWP + 255) / 256, 256, 0, stream>>>(Wf, Wb, Wt);
    kConvU<<<(2 * NG * KUP + 255) / 256, 256, 0, stream>>>(Uf, Ub, Ut);
    kMask<<<(BB * SS) / 8, 256, 0, stream>>>(x, mask);

    size_t smemG = (size_t)64 * SA_STRIDE * 2 + 800 * 144;        // 41,984 + 115,200
    kInputGemm<<<dim3(SS, 5, 2), 256, smemG, stream>>>(x, Wt, bf, bb, xz);

    size_t smemR = (size_t)16 * KUP * 2 + (size_t)48 * NG * 4;    // 7,168 + 3*51,200
    kRecurrent<<<dim3(4, 2), 320, smemR, stream>>>(xz, Ut, mask, hseq);

    kAtt<<<BB, 256, 0, stream>>>(hseq, watt, out);
}
